// SenderDecoder_56169582297445
// MI455X (gfx1250) — matrix operand order repaired and checked by reading, not compiled, not run on hardware
//
#include <hip/hip_runtime.h>
#include <hip/hip_bf16.h>
#include <math.h>

// Problem constants (from reference)
#define BB 512
#define SS 128
#define HH 1024
#define VV 4096
#define TT 32
#define KV 129          // S+1 attention keys
#define H3 3072         // 3*H

typedef __attribute__((ext_vector_type(16))) __bf16       v16bf;
typedef __attribute__((ext_vector_type(8)))  float        v8f;
typedef __attribute__((ext_vector_type(8)))  unsigned int v8u;

__device__ __forceinline__ unsigned short bf16_rne(float x) {
    unsigned int u = __float_as_uint(x);
    u += 0x7FFFu + ((u >> 16) & 1u);
    return (unsigned short)(u >> 16);
}

// CDNA5 async copy: 16 bytes global -> LDS, tracked by ASYNCcnt
__device__ __forceinline__ void async_copy16(unsigned lds_addr, const void* gptr) {
    unsigned long long ga = (unsigned long long)gptr;
    asm volatile("global_load_async_to_lds_b128 %0, %1, off"
                 :: "v"(lds_addr), "v"(ga) : "memory");
}

// ---------------- prep kernels ----------------

__global__ __launch_bounds__(256) void f32_to_bf16_kernel(
    const float* __restrict__ in, unsigned short* __restrict__ out, int n) {
    int i = blockIdx.x * 256 + threadIdx.x;
    if (i < n) out[i] = bf16_rne(in[i]);
}

__global__ __launch_bounds__(256) void init_state_kernel(
    float* __restrict__ h, unsigned short* __restrict__ hbf,
    int* __restrict__ tok,
    float* __restrict__ seq, float* __restrict__ lps, float* __restrict__ ent) {
    int i = blockIdx.x * 256 + threadIdx.x;              // over B*H
    h[i] = 0.f;
    hbf[i] = 0;
    if (i < BB) {
        tok[i] = VV + 2;                                  // SOS token
        seq[i * (TT + 1) + TT] = 0.f;                     // appended zero column
        lps[i * (TT + 1) + TT] = 0.f;
        ent[i * (TT + 1) + TT] = 0.f;
    }
}

// ---------------- per-step kernels ----------------

// x_bf[b, :] = emb_bf[tok[b], :]
__global__ __launch_bounds__(256) void gather_kernel(
    const int* __restrict__ tok, const unsigned short* __restrict__ embb,
    unsigned short* __restrict__ xbf) {
    int i = blockIdx.x * 256 + threadIdx.x;              // over B*H
    int b = i >> 10, j = i & (HH - 1);
    xbf[i] = embb[(size_t)tok[b] * HH + j];
}

// C[M,N] = A[M,K](bf16) * W[N,K](bf16)^T (+ bias), via v_wmma_f32_16x16x32_bf16.
// Block = 256 threads (8 waves) -> 128(M) x 64(N) tile; the 64-column weight
// strip is staged in LDS with double-buffered global_load_async_to_lds_b128
// (ASYNCcnt) and shared by all 8 waves. All fragment loads for a k-step are
// issued before the 4 WMMAs so partial dscnt waits hide LDS latency.
// grid: (N/64, M/128).
#define KS 64                 // K-slice per LDS stage (2 WMMA k-steps)
#define LPW 36                // padded LDS row stride in dwords (144 B)
#define TILE_DW (64 * LPW)    // dwords per buffer

__global__ __launch_bounds__(256) void gemm_bf16_kernel(
    const unsigned short* __restrict__ A, const unsigned short* __restrict__ W,
    float* __restrict__ C, const float* __restrict__ bias,
    int N, int K) {
    __shared__ unsigned int btile[2 * TILE_DW];           // 18 KB

    const int tid  = threadIdx.x;
    const int lane = tid & 31;
    const int wave = tid >> 5;
    const int half = lane >> 4;        // K-half selector (A/B frag layout)
    const int l16  = lane & 15;
    const int mT   = blockIdx.y * 128 + wave * 16;
    const int nT   = blockIdx.x * 64;

    const unsigned int* __restrict__ Au = (const unsigned int*)A;
    const int K2 = K >> 1;
    const int NS = K / KS;

    const unsigned ldsBase = (unsigned)(unsigned long long)(const void*)btile;
    const char* Wb = (const char*)W;

    // per-thread async-copy assignment: 2 x 16B chunks per slice
    const int nl0 = tid >> 3,          c0 = tid & 7;
    const int nl1 = (tid + 256) >> 3,  c1 = tid & 7;

    auto stage = [&](int s, int buf) {
        const size_t k0b = (size_t)s * (KS * 2);          // byte offset into K
        async_copy16(ldsBase + buf * (TILE_DW * 4) + nl0 * 144 + c0 * 16,
                     Wb + (size_t)(nT + nl0) * (K * 2) + k0b + c0 * 16);
        async_copy16(ldsBase + buf * (TILE_DW * 4) + nl1 * 144 + c1 * 16,
                     Wb + (size_t)(nT + nl1) * (K * 2) + k0b + c1 * 16);
    };

    // A-fragment dword offsets within a 32-wide K slice (ISA 16-bit A layout)
    int akb2[8];
#pragma unroll
    for (int i = 0; i < 8; ++i) akb2[i] = ((i < 4) ? i : i + 4) + 4 * half;
    const size_t aBase = (size_t)(mT + l16) * K2;

    v8f acc[4] = {v8f{}, v8f{}, v8f{}, v8f{}};

    stage(0, 0);
    if (NS > 1) stage(1, 1);

    for (int s = 0; s < NS; ++s) {
        if (s + 1 < NS) asm volatile("s_wait_asynccnt 0x2" ::: "memory");
        else            asm volatile("s_wait_asynccnt 0x0" ::: "memory");
        __syncthreads();                                   // slice s visible to all

        const unsigned int* bt = btile + (s & 1) * TILE_DW;
#pragma unroll
        for (int kk = 0; kk < KS; kk += 32) {
            // ---- issue ALL fragment loads for this k-step first ----
            v8u au;
#pragma unroll
            for (int i = 0; i < 8; ++i)
                au[i] = Au[aBase + ((s * KS + kk) >> 1) + akb2[i]];

            v8u bu[4];
#pragma unroll
            for (int j = 0; j < 4; ++j) {
                const int base = (j * 16 + l16) * LPW + (kk >> 1) + half * 4;
#pragma unroll
                for (int i = 0; i < 8; ++i) bu[j][i] = bt[base + ((i < 4) ? i : i + 4)];
            }

            // ---- then 4 back-to-back WMMAs (partial waits hide latency) ----
            v16bf av = __builtin_bit_cast(v16bf, au);
#pragma unroll
            for (int j = 0; j < 4; ++j) {
                v16bf bv = __builtin_bit_cast(v16bf, bu[j]);
                acc[j] = __builtin_amdgcn_wmma_f32_16x16x32_bf16(
                    false, av, false, bv, (short)0, acc[j], false, false);
            }
        }

        __syncthreads();                                   // buffer free to refill
        if (s + 2 < NS) stage(s + 2, s & 1);
    }

#pragma unroll
    for (int j = 0; j < 4; ++j) {
        const int n = nT + j * 16 + l16;
        const float bz = bias ? bias[n] : 0.f;
#pragma unroll
        for (int i = 0; i < 8; ++i) {
            const int row = mT + i + 8 * half;            // ISA C/D layout
            C[(size_t)row * N + n] = acc[j][i] + bz;
        }
    }
}

// GRU gates: r,z,n (PyTorch order), updates h (f32) and hbf (bf16)
__global__ __launch_bounds__(256) void gru_gates_kernel(
    const float* __restrict__ gi, const float* __restrict__ gh,
    const float* __restrict__ b_ih, const float* __restrict__ b_hh,
    float* __restrict__ h, unsigned short* __restrict__ hbf) {
    int i = blockIdx.x * 256 + threadIdx.x;              // over B*H
    int b = i >> 10, j = i & (HH - 1);
    size_t g = (size_t)b * H3 + j;
    float ir = gi[g]            + b_ih[j];
    float iz = gi[g + HH]       + b_ih[HH + j];
    float in_ = gi[g + 2 * HH]  + b_ih[2 * HH + j];
    float hr = gh[g]            + b_hh[j];
    float hz = gh[g + HH]       + b_hh[HH + j];
    float hn = gh[g + 2 * HH]   + b_hh[2 * HH + j];
    float r = 1.f / (1.f + expf(-(ir + hr)));
    float z = 1.f / (1.f + expf(-(iz + hz)));
    float n = tanhf(in_ + r * hn);
    float hnew = (1.f - z) * n + z * h[i];
    h[i] = hnew;
    hbf[i] = bf16_rne(hnew);
}

// scores[b,k] = (1/32) * dot(h[b,:], keys[b,k,:]); keys row 128 = embedding[V]
__global__ __launch_bounds__(256) void attn_scores_kernel(
    const float* __restrict__ h, const float* __restrict__ enc,
    const float* __restrict__ emb, float* __restrict__ sc) {
    __shared__ float red[256];
    const int k = blockIdx.x, b = blockIdx.y, tid = threadIdx.x;
    const float* key = (k < SS) ? (enc + ((size_t)b * SS + k) * HH)
                                : (emb + (size_t)VV * HH);
    const float* hb = h + (size_t)b * HH;
    float s = 0.f;
#pragma unroll
    for (int j = tid; j < HH; j += 256) s += hb[j] * key[j];
    red[tid] = s;
    __syncthreads();
    for (int st = 128; st > 0; st >>= 1) {
        if (tid < st) red[tid] += red[tid + st];
        __syncthreads();
    }
    if (tid == 0) sc[(size_t)b * KV + k] = red[0] * 0.03125f;  // 1/sqrt(1024)
}

// softmax over 129; writes normalized w back to sc and to output ws slice
__global__ __launch_bounds__(256) void attn_softmax_kernel(
    float* __restrict__ sc, float* __restrict__ wout, int t) {
    __shared__ float red[256];
    const int b = blockIdx.x, tid = threadIdx.x;
    float v = (tid < KV) ? sc[(size_t)b * KV + tid] : -INFINITY;
    red[tid] = v;
    __syncthreads();
    for (int st = 128; st > 0; st >>= 1) {
        if (tid < st) red[tid] = fmaxf(red[tid], red[tid + st]);
        __syncthreads();
    }
    const float m = red[0];
    __syncthreads();
    float e = (tid < KV) ? expf(v - m) : 0.f;
    red[tid] = e;
    __syncthreads();
    for (int st = 128; st > 0; st >>= 1) {
        if (tid < st) red[tid] += red[tid + st];
        __syncthreads();
    }
    const float invZ = 1.f / red[0];
    if (tid < KV) {
        float w = e * invZ;
        sc[(size_t)b * KV + tid] = w;
        wout[((size_t)b * TT + t) * KV + tid] = w;
    }
}

// rh_bf[b,j] = bf16( h[b,j] + sum_k w[b,k]*values[b,k,j] ); values row 128 = embedding[V+1]
__global__ __launch_bounds__(256) void attn_context_kernel(
    const float* __restrict__ sc, const float* __restrict__ sem,
    const float* __restrict__ emb, const float* __restrict__ h,
    unsigned short* __restrict__ rhbf) {
    __shared__ float wsh[KV];
    const int b = blockIdx.y, tid = threadIdx.x;
    const int j = blockIdx.x * 256 + tid;
    if (tid < KV) wsh[tid] = sc[(size_t)b * KV + tid];
    __syncthreads();
    float acc = 0.f;
    const float* sb = sem + (size_t)b * SS * HH + j;
#pragma unroll 4
    for (int k = 0; k < SS; ++k) {
        if (k + 8 < SS) __builtin_prefetch(sb + (size_t)(k + 8) * HH, 0, 1);
        acc += wsh[k] * sb[(size_t)k * HH];
    }
    acc += wsh[SS] * emb[(size_t)(VV + 1) * HH + j];
    rhbf[(size_t)b * HH + j] = bf16_rne(acc + h[(size_t)b * HH + j]);
}

// per-row argmax + log-softmax stats over V=4096
__global__ __launch_bounds__(256) void argmax_kernel(
    const float* __restrict__ logits, int* __restrict__ tok,
    float* __restrict__ seq, float* __restrict__ lps, float* __restrict__ ent,
    int t) {
    __shared__ float smax[256];
    __shared__ int   sidx[256];
    __shared__ float ssum[256];
    __shared__ float ssum2[256];
    const int b = blockIdx.x, tid = threadIdx.x;
    const float* lb = logits + (size_t)b * VV;

    float lm = -INFINITY; int li = 0;
    for (int v = tid; v < VV; v += 256) {
        float l = lb[v];
        if (l > lm) { lm = l; li = v; }
    }
    smax[tid] = lm; sidx[tid] = li;
    __syncthreads();
    for (int st = 128; st > 0; st >>= 1) {
        if (tid < st) {
            float a = smax[tid], c = smax[tid + st];
            int ai = sidx[tid], ci = sidx[tid + st];
            if (c > a || (c == a && ci < ai)) { smax[tid] = c; sidx[tid] = ci; }
        }
        __syncthreads();
    }
    const float m = smax[0];
    const int am = sidx[0];
    __syncthreads();

    float z = 0.f, s2 = 0.f;
    for (int v = tid; v < VV; v += 256) {
        float d = lb[v] - m;
        float e = expf(d);
        z += e; s2 += e * d;
    }
    ssum[tid] = z; ssum2[tid] = s2;
    __syncthreads();
    for (int st = 128; st > 0; st >>= 1) {
        if (tid < st) { ssum[tid] += ssum[tid + st]; ssum2[tid] += ssum2[tid + st]; }
        __syncthreads();
    }
    if (tid == 0) {
        float Z = ssum[0];
        float logZ = logf(Z);
        tok[b] = am;
        seq[(size_t)b * (TT + 1) + t] = (float)am;
        lps[(size_t)b * (TT + 1) + t] = -logZ;                 // logp at argmax
        ent[(size_t)b * (TT + 1) + t] = logZ - ssum2[0] / Z;   // entropy
    }
}

// ---------------- host ----------------

extern "C" void kernel_launch(void* const* d_in, const int* in_sizes, int n_in,
                              void* d_out, int out_size, void* d_ws, size_t ws_size,
                              hipStream_t stream) {
    (void)in_sizes; (void)n_in; (void)out_size; (void)ws_size;

    const float* enc   = (const float*)d_in[0];   // [B,S,H]
    const float* sem   = (const float*)d_in[1];   // [B,S,H]
    const float* emb   = (const float*)d_in[2];   // [V+3,H]
    const float* W_ih  = (const float*)d_in[3];   // [3H,H]
    const float* W_hh  = (const float*)d_in[4];   // [3H,H]
    const float* b_ih  = (const float*)d_in[5];   // [3H]
    const float* b_hh  = (const float*)d_in[6];   // [3H]
    const float* W_out = (const float*)d_in[7];   // [V,H]
    const float* b_out = (const float*)d_in[8];   // [V]

    // workspace layout (256B aligned slabs)
    char* wsb = (char*)d_ws;
    size_t off = 0;
    auto take = [&](size_t bytes) -> void* {
        void* p = wsb + off;
        off = (off + bytes + 255) & ~(size_t)255;
        return p;
    };
    float*          h      = (float*)take((size_t)BB * HH * 4);
    unsigned short* hbf    = (unsigned short*)take((size_t)BB * HH * 2);
    unsigned short* xbf    = (unsigned short*)take((size_t)BB * HH * 2);
    float*          gi     = (float*)take((size_t)BB * H3 * 4);
    float*          gh     = (float*)take((size_t)BB * H3 * 4);
    unsigned short* rhbf   = (unsigned short*)take((size_t)BB * HH * 2);
    float*          sc     = (float*)take((size_t)BB * KV * 4);
    float*          logits = (float*)take((size_t)BB * VV * 4);
    int*            tok    = (int*)take((size_t)BB * 4);
    unsigned short* Wihb   = (unsigned short*)take((size_t)H3 * HH * 2);
    unsigned short* Whhb   = (unsigned short*)take((size_t)H3 * HH * 2);
    unsigned short* Woutb  = (unsigned short*)take((size_t)VV * HH * 2);
    unsigned short* embb   = (unsigned short*)take((size_t)(VV + 3) * HH * 2);

    // output layout: seq [B,T+1] | lps [B,T+1] | ent [B,T+1] | w [B,T,129]
    float* seq  = (float*)d_out;
    float* lps  = seq + (size_t)BB * (TT + 1);
    float* ent  = lps + (size_t)BB * (TT + 1);
    float* wout = ent + (size_t)BB * (TT + 1);

    // one-time (per launch) prep: bf16 weight copies + state init
    {
        int n;
        n = H3 * HH;
        f32_to_bf16_kernel<<<(n + 255) / 256, 256, 0, stream>>>(W_ih, Wihb, n);
        f32_to_bf16_kernel<<<(n + 255) / 256, 256, 0, stream>>>(W_hh, Whhb, n);
        n = VV * HH;
        f32_to_bf16_kernel<<<(n + 255) / 256, 256, 0, stream>>>(W_out, Woutb, n);
        n = (VV + 3) * HH;
        f32_to_bf16_kernel<<<(n + 255) / 256, 256, 0, stream>>>(emb, embb, n);
        init_state_kernel<<<(BB * HH) / 256, 256, 0, stream>>>(h, hbf, tok, seq, lps, ent);
    }

    const dim3 gGemm3(H3 / 64, BB / 128);   // [512,3072] GEMMs
    const dim3 gGemmV(VV / 64, BB / 128);   // [512,4096] GEMM
    const dim3 gScore(KV, BB);
    const dim3 gCtx(HH / 256, BB);

    for (int t = 0; t < TT; ++t) {
        gather_kernel<<<(BB * HH) / 256, 256, 0, stream>>>(tok, embb, xbf);
        gemm_bf16_kernel<<<gGemm3, 256, 0, stream>>>(xbf, Wihb, gi, nullptr, H3, HH);
        gemm_bf16_kernel<<<gGemm3, 256, 0, stream>>>(hbf, Whhb, gh, nullptr, H3, HH);
        gru_gates_kernel<<<(BB * HH) / 256, 256, 0, stream>>>(gi, gh, b_ih, b_hh, h, hbf);
        attn_scores_kernel<<<gScore, 256, 0, stream>>>(h, enc, emb, sc);
        attn_softmax_kernel<<<BB, 256, 0, stream>>>(sc, wout, t);
        attn_context_kernel<<<gCtx, 256, 0, stream>>>(sc, sem, emb, h, rhbf);
        gemm_bf16_kernel<<<gGemmV, 256, 0, stream>>>(rhbf, Woutb, logits, b_out, VV, HH);
        argmax_kernel<<<BB, 256, 0, stream>>>(logits, tok, seq, lps, ent, t);
    }
}
